// Decoder_206158430399
// MI455X (gfx1250) — matrix operand order repaired and checked by reading, not compiled, not run on hardware
//
#include <hip/hip_runtime.h>
#include <hip/hip_bf16.h>

typedef __attribute__((ext_vector_type(16))) __bf16 v16bf;
typedef __attribute__((ext_vector_type(8)))  __bf16 v8bf;
typedef __attribute__((ext_vector_type(8)))  float  v8f;

#define BB 128
#define NN 256
#define HH 512
#define TT 32

// ---------------------------------------------------------------------------
// Generic GEMM: out[M x Nn] = A[M x K] @ W[Nn x K]^T (+ bias[Nn])
// bf16 inputs, fp32 accumulate via v_wmma_f32_16x16x32_bf16.
// Block = 256 threads = 8 waves. Each wave computes a 16x64 tile:
//   mTile = blockIdx.y*8 + wave ; n range = blockIdx.x*64 .. +63 (4 n-tiles).
// ---------------------------------------------------------------------------
__global__ __launch_bounds__(256) void gemm_bf16_wmma(
    const __bf16* __restrict__ A, const __bf16* __restrict__ W,
    const float* __restrict__ bias, float* __restrict__ out,
    int M, int Nn, int K)
{
    const int wave  = threadIdx.x >> 5;
    const int lane  = threadIdx.x & 31;
    const int lhalf = lane >> 4;   // 0 or 1
    const int lmod  = lane & 15;

    const int m0 = (blockIdx.y * 8 + wave) * 16;
    if (m0 >= M) return;
    const int n0 = blockIdx.x * 64;

    const __bf16* __restrict__ arow = A + (long)(m0 + lmod) * K;
    const __bf16* __restrict__ br0  = W + (long)(n0 +  0 + lmod) * K;
    const __bf16* __restrict__ br1  = W + (long)(n0 + 16 + lmod) * K;
    const __bf16* __restrict__ br2  = W + (long)(n0 + 32 + lmod) * K;
    const __bf16* __restrict__ br3  = W + (long)(n0 + 48 + lmod) * K;

    v8f acc0 = {}, acc1 = {}, acc2 = {}, acc3 = {};

    for (int k = 0; k < K; k += 32) {
        // A fragment: row m0+lmod, 16 contiguous bf16 at k + lhalf*16
        v8bf alo = *(const v8bf*)(arow + k + lhalf * 16);
        v8bf ahi = *(const v8bf*)(arow + k + lhalf * 16 + 8);
        v16bf af;
        #pragma unroll
        for (int i = 0; i < 8; ++i) { af[i] = alo[i]; af[8 + i] = ahi[i]; }

        // B fragments: row n0+sub*16+lmod, 16 contiguous bf16 at k + lhalf*16
        v16bf b0 = *(const v16bf*)(br0 + k + lhalf * 16);
        v16bf b1 = *(const v16bf*)(br1 + k + lhalf * 16);
        v16bf b2 = *(const v16bf*)(br2 + k + lhalf * 16);
        v16bf b3 = *(const v16bf*)(br3 + k + lhalf * 16);

        acc0 = __builtin_amdgcn_wmma_f32_16x16x32_bf16(false, af, false, b0, (short)0, acc0, false, false);
        acc1 = __builtin_amdgcn_wmma_f32_16x16x32_bf16(false, af, false, b1, (short)0, acc1, false, false);
        acc2 = __builtin_amdgcn_wmma_f32_16x16x32_bf16(false, af, false, b2, (short)0, acc2, false, false);
        acc3 = __builtin_amdgcn_wmma_f32_16x16x32_bf16(false, af, false, b3, (short)0, acc3, false, false);
    }

    // C/D layout: acc[j] -> row m0 + 8*lhalf + j, col n0 + sub*16 + lmod
    const int rbase = m0 + 8 * lhalf;
    #pragma unroll
    for (int j = 0; j < 8; ++j) {
        long r = rbase + j;
        int c0 = n0 +  0 + lmod, c1 = n0 + 16 + lmod;
        int c2 = n0 + 32 + lmod, c3 = n0 + 48 + lmod;
        float b0f = bias ? bias[c0] : 0.f, b1f = bias ? bias[c1] : 0.f;
        float b2f = bias ? bias[c2] : 0.f, b3f = bias ? bias[c3] : 0.f;
        out[r * Nn + c0] = acc0[j] + b0f;
        out[r * Nn + c1] = acc1[j] + b1f;
        out[r * Nn + c2] = acc2[j] + b2f;
        out[r * Nn + c3] = acc3[j] + b3f;
    }
}

// ---------------------------------------------------------------------------
// Elementwise helpers
// ---------------------------------------------------------------------------
__global__ void f32_to_bf16(const float* __restrict__ src, __bf16* __restrict__ dst, long n)
{
    long i = (long)blockIdx.x * blockDim.x + threadIdx.x;
    if (i < n) dst[i] = (__bf16)src[i];
}

__global__ void zero_u32(unsigned int* __restrict__ p, long n)
{
    long i = (long)blockIdx.x * blockDim.x + threadIdx.x;
    if (i < n) p[i] = 0u;
}

// GRU gates: r/z/n from gi,gh; update h in place; also emit bf16 copy of h.
__global__ void gru_gates(const float* __restrict__ gi, const float* __restrict__ gh,
                          float* __restrict__ h, __bf16* __restrict__ h_bf)
{
    int i = blockIdx.x * blockDim.x + threadIdx.x;   // over B*H
    int b = i / HH, j = i % HH;
    const float* gib = gi + (long)b * 3 * HH;
    const float* ghb = gh + (long)b * 3 * HH;
    float r = 1.f / (1.f + __expf(-(gib[j] + ghb[j])));
    float z = 1.f / (1.f + __expf(-(gib[HH + j] + ghb[HH + j])));
    float nn = tanhf(gib[2 * HH + j] + r * ghb[2 * HH + j]);
    float hv = (1.f - z) * nn + z * h[i];
    h[i] = hv;
    h_bf[i] = (__bf16)hv;
}

// Attention scores: u[row] = sum_h tanh(W1e[row,h] + W2h[b,h]) * v[h]
// 8 waves per block, one (b,n) row per wave. W1e is L2-resident (64MB < 192MB).
__global__ __launch_bounds__(256) void attn_scores(
    const float* __restrict__ W1e, const float* __restrict__ W2h,
    const float* __restrict__ vvec, float* __restrict__ u)
{
    int lane = threadIdx.x & 31;
    int wave = threadIdx.x >> 5;
    long row = (long)blockIdx.x * 8 + wave;          // b*N + n
    int  b   = (int)(row / NN);
    const float* w1 = W1e + row * HH;
    const float* w2 = W2h + (long)b * HH;
    float acc = 0.f;
    #pragma unroll 4
    for (int t = 0; t < HH / 32; ++t) {
        int i = t * 32 + lane;
        acc += tanhf(w1[i] + w2[i]) * vvec[i];
    }
    #pragma unroll
    for (int o = 16; o; o >>= 1) acc += __shfl_xor(acc, o, 32);
    if (lane == 0) u[row] = acc;
}

// Per-step softmax / argmax / logp / entropy / mask update / embed gather.
// One block per batch row b; 256 threads, thread n handles element n.
__global__ __launch_bounds__(256) void select_step(
    const float* __restrict__ u, int* __restrict__ mask,
    const __bf16* __restrict__ enc_bf, __bf16* __restrict__ prev_bf,
    float* __restrict__ out, int t)
{
    __shared__ float sval[NN];
    __shared__ int   sidx[NN];
    int b = blockIdx.x, n = threadIdx.x;

    float val = mask[b * NN + n] ? -1e9f : u[b * NN + n];

    // argmax (first index on ties, to match jnp.argmax)
    sval[n] = val; sidx[n] = n; __syncthreads();
    for (int s = NN / 2; s; s >>= 1) {
        if (n < s) {
            float ov = sval[n + s]; int oi = sidx[n + s];
            if (ov > sval[n] || (ov == sval[n] && oi < sidx[n])) { sval[n] = ov; sidx[n] = oi; }
        }
        __syncthreads();
    }
    float vmax = sval[0]; int idx = sidx[0]; __syncthreads();

    float e = __expf(val - vmax);
    sval[n] = e; __syncthreads();
    for (int s = NN / 2; s; s >>= 1) { if (n < s) sval[n] += sval[n + s]; __syncthreads(); }
    float denom = sval[0]; __syncthreads();

    float p = e / denom;
    sval[n] = -p * __logf(p + 1e-9f); __syncthreads();
    for (int s = NN / 2; s; s >>= 1) { if (n < s) sval[n] += sval[n + s]; __syncthreads(); }
    float ent = sval[0];

    if (n == 0) {
        out[0 * BB * TT + b * TT + t] = (float)idx;                    // idx
        out[1 * BB * TT + b * TT + t] = __logf(1.f / denom + 1e-9f);   // logp (p[idx]=1/denom)
        out[2 * BB * TT + b * TT + t] = ent;                           // entropy
        mask[b * NN + idx] = 1;
    }

    // gather next prev_embed (bf16 for the GRU WMMA GEMM)
    for (int i = n; i < HH; i += NN)
        prev_bf[(long)b * HH + i] = enc_bf[((long)b * NN + idx) * HH + i];
}

// ---------------------------------------------------------------------------
extern "C" void kernel_launch(void* const* d_in, const int* in_sizes, int n_in,
                              void* d_out, int out_size, void* d_ws, size_t ws_size,
                              hipStream_t stream) {
    const float* enc  = (const float*)d_in[0];   // B*N*H
    const float* w_ih = (const float*)d_in[1];   // 3H*H
    const float* w_hh = (const float*)d_in[2];   // 3H*H
    const float* b_ih = (const float*)d_in[3];   // 3H
    const float* b_hh = (const float*)d_in[4];   // 3H
    const float* W1   = (const float*)d_in[5];   // H*H
    const float* W2   = (const float*)d_in[6];   // H*H
    const float* vvec = (const float*)d_in[7];   // H
    float* out = (float*)d_out;

    char* ws = (char*)d_ws;
    const long ENCN = (long)BB * NN * HH;        // 16,777,216
    size_t off = 0;
    auto alloc = [&](size_t bytes) { void* p = ws + off; off = (off + bytes + 255) & ~(size_t)255; return p; };

    __bf16* enc_bf  = (__bf16*)alloc(ENCN * 2);
    __bf16* W1_bf   = (__bf16*)alloc((size_t)HH * HH * 2);
    __bf16* W2_bf   = (__bf16*)alloc((size_t)HH * HH * 2);
    __bf16* wih_bf  = (__bf16*)alloc((size_t)3 * HH * HH * 2);
    __bf16* whh_bf  = (__bf16*)alloc((size_t)3 * HH * HH * 2);
    float*  W1e     = (float*)alloc(ENCN * 4);
    float*  h       = (float*)alloc((size_t)BB * HH * 4);
    __bf16* h_bf    = (__bf16*)alloc((size_t)BB * HH * 2);
    __bf16* prev_bf = (__bf16*)alloc((size_t)BB * HH * 2);
    float*  gi      = (float*)alloc((size_t)BB * 3 * HH * 4);
    float*  gh      = (float*)alloc((size_t)BB * 3 * HH * 4);
    float*  W2h     = (float*)alloc((size_t)BB * HH * 4);
    float*  u       = (float*)alloc((size_t)BB * NN * 4);
    int*    mask    = (int*)alloc((size_t)BB * NN * 4);

    // 1) convert inputs to bf16
    f32_to_bf16<<<dim3((unsigned)((ENCN + 255) / 256)), 256, 0, stream>>>(enc, enc_bf, ENCN);
    f32_to_bf16<<<dim3((HH * HH + 255) / 256), 256, 0, stream>>>(W1, W1_bf, (long)HH * HH);
    f32_to_bf16<<<dim3((HH * HH + 255) / 256), 256, 0, stream>>>(W2, W2_bf, (long)HH * HH);
    f32_to_bf16<<<dim3((3 * HH * HH + 255) / 256), 256, 0, stream>>>(w_ih, wih_bf, (long)3 * HH * HH);
    f32_to_bf16<<<dim3((3 * HH * HH + 255) / 256), 256, 0, stream>>>(w_hh, whh_bf, (long)3 * HH * HH);

    // 2) zero state: h (fp32), h_bf + prev_bf (bf16), mask
    zero_u32<<<dim3((BB * HH + 255) / 256), 256, 0, stream>>>((unsigned*)h, BB * HH);
    zero_u32<<<dim3((BB * HH / 2 + 255) / 256), 256, 0, stream>>>((unsigned*)h_bf, BB * HH / 2);
    zero_u32<<<dim3((BB * HH / 2 + 255) / 256), 256, 0, stream>>>((unsigned*)prev_bf, BB * HH / 2);
    zero_u32<<<dim3((BB * NN + 255) / 256), 256, 0, stream>>>((unsigned*)mask, BB * NN);

    // 3) loop-invariant W1e = enc @ W1^T : M=32768, N=512, K=512
    gemm_bf16_wmma<<<dim3(HH / 64, (BB * NN) / 128), 256, 0, stream>>>(
        enc_bf, W1_bf, nullptr, W1e, BB * NN, HH, HH);

    // 4) sequential decode (decode_steps = 32 per setup_inputs; compiled in)
    for (int t = 0; t < TT; ++t) {
        // gi = prev_embed @ w_ih^T + b_ih ; gh = h @ w_hh^T + b_hh (both use pre-update state)
        gemm_bf16_wmma<<<dim3(3 * HH / 64, 1), 256, 0, stream>>>(prev_bf, wih_bf, b_ih, gi, BB, 3 * HH, HH);
        gemm_bf16_wmma<<<dim3(3 * HH / 64, 1), 256, 0, stream>>>(h_bf,    whh_bf, b_hh, gh, BB, 3 * HH, HH);
        gru_gates<<<dim3(BB * HH / 256), 256, 0, stream>>>(gi, gh, h, h_bf);
        // W2h = h_new @ W2^T
        gemm_bf16_wmma<<<dim3(HH / 64, 1), 256, 0, stream>>>(h_bf, W2_bf, nullptr, W2h, BB, HH, HH);
        // u[b,n] = tanh(W1e + W2h) . v   (W1e streams from L2)
        attn_scores<<<dim3(BB * NN / 8), 256, 0, stream>>>(W1e, W2h, vvec, u);
        // softmax/argmax/logp/entropy + mask update + embed gather
        select_step<<<dim3(BB), NN, 0, stream>>>(u, mask, enc_bf, prev_bf, out, t);
    }
}
